// LocalCrossAttention1D_54589034332797
// MI455X (gfx1250) — matrix operand order repaired and checked by reading, not compiled, not run on hardware
//
#include <hip/hip_runtime.h>
#include <cstdint>
#include <cstddef>

// ---------------------------------------------------------------------------
// LocalCrossAttention1D for MI455X (gfx1250, wave32, WMMA)
// All matmuls via v_wmma_f32_16x16x32_f16 (f16 in, f32 accumulate).
// GEMM stages activations into LDS with global_load_async_to_lds_b128.
// ---------------------------------------------------------------------------

typedef __attribute__((ext_vector_type(16))) _Float16 v16h;
typedef __attribute__((ext_vector_type(8)))  _Float16 v8h;
typedef __attribute__((ext_vector_type(8)))  float    v8f;

#define EPS_IN 1e-5f

// Problem constants (fixed by setup_inputs)
#define BB   4
#define CDIM 512
#define CE   256
#define LDEC 4096
#define LENC 8192
#define HDIM 2048
#define NH   8
#define HD   64
#define WIN  256
#define STR  192
#define NWIN 22     // ceil(4096/192)
#define MAXE 512    // max encoder window length

// --- WMMA fragment index helpers (ISA 7.12.2, 16-bit data, wave32) ---------
// A matrix 16x32 (MxK): M = lane&15; elem e: K = 16*(e>>3) + 8*(lane>>4) + (e&7)
//   => elems 0..7  : K = 8*(lane>>4) + 0..7        (contiguous, 16B)
//      elems 8..15 : K = 16 + 8*(lane>>4) + 0..7   (contiguous, 16B)
__device__ __forceinline__ int a_k(int lane, int e) {
  return ((e >> 3) << 4) + ((lane >> 4) << 3) + (e & 7);
}
// B matrix 32x16 (KxN): N = lane&15; K = 16*(e>>3) + 8*(lane>>4) + (e&7)
__device__ __forceinline__ int b_k(int lane, int e) {
  return ((e >> 3) << 4) + ((lane >> 4) << 3) + (e & 7);
}
// C/D 16x16 f32: elem p: M = p + 8*(lane>>4), N = lane&15

__device__ __forceinline__ v8f wmma_f16(v16h a, v16h b, v8f c) {
  return __builtin_amdgcn_wmma_f32_16x16x32_f16(false, a, false, b,
                                                (short)0, c, false, false);
}
__device__ __forceinline__ v16h cat8(v8h lo, v8h hi) {
  return __builtin_shufflevector(lo, hi, 0, 1, 2, 3, 4, 5, 6, 7,
                                         8, 9, 10, 11, 12, 13, 14, 15);
}

// --- CDNA5 async global->LDS copy (ASYNCcnt path, ISA 10.x / 08_async) -----
// Flat pointers to LDS carry the LDS offset in addr[31:0] (aperture in the
// high half), so truncating a generic pointer yields the LDS byte address.
__device__ __forceinline__ void async_load_b128(const void* sbase, int goff,
                                                uint32_t lds_addr) {
  asm volatile("global_load_async_to_lds_b128 %0, %1, %2"
               :: "v"(lds_addr), "v"(goff), "s"(sbase)
               : "memory");
}
__device__ __forceinline__ void wait_async0() {
  asm volatile("s_wait_asynccnt 0" ::: "memory");
}

// ---------------------------------------------------------------------------
// conv1x1 GEMM: Y[b,o,l] = sum_c W[o,c] * X[b,c,l] + bias[o]
// flags: bit0 = accumulate per-(b,o) sum/sumsq into stats (instance norm)
//        bit1 = apply exact GELU to output
// Block 256 thr = 8 waves; block tile 128(M) x 64(N); K-step 32.
// Wave w computes rows [w*16, w*16+16) x all 64 N columns (4 WMMA accs),
// reusing its A fragment 4x.  X tile comes in via async b128 -> LDS.
// grid (L/64, O/128, B)
// ---------------------------------------------------------------------------
__global__ __launch_bounds__(256)
void gemm_wmma(const float* __restrict__ W, const float* __restrict__ X,
               const float* __restrict__ bias, float* __restrict__ Y,
               float* __restrict__ stats, int O, int C, int L, int flags)
{
  __shared__ float    Xs32[32][64];        // raw f32 X tile ([k][n]) via async
  __shared__ _Float16 Xs16[64][32 + 8];    // f16, transposed to [n][k]
  __shared__ _Float16 Ws16[128][32 + 8];   // f16 W tile, [m][k]

  const int t    = threadIdx.x;
  const int lane = t & 31;
  const int wv   = t >> 5;
  const int b    = blockIdx.z;
  const int mBlk = blockIdx.y * 128;
  const int nBlk = blockIdx.x * 64;
  const int hi   = lane >> 4;              // lane-half select
  const int lo   = lane & 15;
  const float* Xb = X + (size_t)b * C * L;

  v8f acc[4] = {};                         // 16 x 64 per wave

  for (int kb = 0; kb < C; kb += 32) {
    __syncthreads();   // protect LDS tiles from previous iteration's readers

    // ---- stage X tile (32k x 64n f32) via async b128: 2 per thread ----
#pragma unroll
    for (int r = 0; r < 2; ++r) {
      int idx = (t + r * 256) * 4;                  // float index 0..2047
      int k = idx >> 6, n = idx & 63;
      int goff = (int)(((size_t)(kb + k) * L + nBlk + n) * sizeof(float));
      async_load_b128(Xb, goff, (uint32_t)(uintptr_t)&Xs32[k][n]);
    }

    // ---- stage W tile (128m x 32k) coalesced f32 -> f16 [m][k] ----
#pragma unroll
    for (int r = 0; r < 4; ++r) {
      int idx = (t + r * 256) * 4;                  // 0..4095
      int m = idx >> 5, c = idx & 31;
      const float4 w4 =
          *reinterpret_cast<const float4*>(&W[(size_t)(mBlk + m) * C + kb + c]);
      Ws16[m][c + 0] = (_Float16)w4.x;
      Ws16[m][c + 1] = (_Float16)w4.y;
      Ws16[m][c + 2] = (_Float16)w4.z;
      Ws16[m][c + 3] = (_Float16)w4.w;
    }

    wait_async0();
    __syncthreads();

    // ---- transpose-convert X tile to f16 [n][k]: 8 per thread ----
#pragma unroll
    for (int r = 0; r < 8; ++r) {
      int idx = t + r * 256;                        // 0..2047
      int k = idx >> 6, n = idx & 63;
      Xs16[n][k] = (_Float16)Xs32[k][n];
    }
    __syncthreads();

    // ---- fragments from LDS (aligned b128 reads) + 4 WMMAs ----
    const int am = wv * 16 + lo;
    v8h a0 = *reinterpret_cast<const v8h*>(&Ws16[am][hi * 8]);
    v8h a1 = *reinterpret_cast<const v8h*>(&Ws16[am][16 + hi * 8]);
    v16h af = cat8(a0, a1);
#pragma unroll
    for (int nt = 0; nt < 4; ++nt) {
      const int n = nt * 16 + lo;
      v8h b0 = *reinterpret_cast<const v8h*>(&Xs16[n][hi * 8]);
      v8h b1 = *reinterpret_cast<const v8h*>(&Xs16[n][16 + hi * 8]);
      acc[nt] = wmma_f16(af, cat8(b0, b1), acc[nt]);
    }
  }

  // ---- epilogue: bias (+gelu) store, optional instance-norm stats ----
#pragma unroll
  for (int nt = 0; nt < 4; ++nt) {
#pragma unroll
    for (int p = 0; p < 8; ++p) {
      const int row = mBlk + wv * 16 + p + (hi << 3);
      const int col = nBlk + nt * 16 + lo;
      float v = acc[nt][p] + bias[row];
      if (flags & 2) v = 0.5f * v * (1.0f + erff(v * 0.70710678118654752f));
      Y[((size_t)b * O + row) * L + col] = v;
      if (flags & 1) {
        float s = v, q = v * v;
#pragma unroll
        for (int m = 8; m >= 1; m >>= 1) {   // reduce across the 16 N-lanes
          s += __shfl_xor(s, m, 32);
          q += __shfl_xor(q, m, 32);
        }
        if (lo == 0) {
          atomicAdd(&stats[((size_t)b * O + row) * 2 + 0], s);
          atomicAdd(&stats[((size_t)b * O + row) * 2 + 1], q);
        }
      }
    }
  }
}

// ---------------------------------------------------------------------------
// Instance-norm finalize: out = (Y - mean)*rsqrt(var+eps) [+ residual]
// ---------------------------------------------------------------------------
__global__ __launch_bounds__(256)
void fin_norm(const float* __restrict__ Y, const float* __restrict__ stats,
              const float* __restrict__ res, float* __restrict__ out,
              int L, size_t total)
{
  size_t i = (size_t)blockIdx.x * blockDim.x + threadIdx.x;
  if (i >= total) return;
  size_t row = i / (size_t)L;            // b*O + o
  float invL = 1.0f / (float)L;
  float m   = stats[row * 2 + 0] * invL;
  float var = stats[row * 2 + 1] * invL - m * m;
  float r   = rsqrtf(var + EPS_IN);
  float v   = (Y[i] - m) * r;
  if (res) v += res[i];
  out[i] = v;
}

__global__ __launch_bounds__(256)
void fill_zero(float* __restrict__ p, size_t n)
{
  size_t i = (size_t)blockIdx.x * blockDim.x + threadIdx.x;
  if (i < n) p[i] = 0.0f;
}

// counts[l] = number of windows whose valid dec range covers l
__global__ void counts_k(float* __restrict__ counts)
{
  int l = blockIdx.x * blockDim.x + threadIdx.x;
  if (l >= LDEC) return;
  float c = 0.0f;
#pragma unroll
  for (int n = 0; n < NWIN; ++n) {
    int s = n * STR;
    int e = min(s + WIN, LDEC);
    if (l >= s && l < e) c += 1.0f;
  }
  counts[l] = c;
}

__global__ __launch_bounds__(256)
void div_counts(float* __restrict__ Oacc, const float* __restrict__ counts, size_t total)
{
  size_t i = (size_t)blockIdx.x * blockDim.x + threadIdx.x;
  if (i >= total) return;
  int l = (int)(i & (LDEC - 1));
  Oacc[i] /= fmaxf(counts[l], 1e-8f);
}

// ---------------------------------------------------------------------------
// Windowed cross-attention, one block = 16 query rows of one window of one
// (b,h).  grid (WIN/16=16, NWIN=22, B*NH=32), block 256 (8 waves).
// Q/K/V layout: (B, Cd, L) with channel = h*HD + d  (so d strides by L).
// Oacc layout:  (B, Cd, Ld) accumulator, atomic scatter-add over overlaps.
// ---------------------------------------------------------------------------
__global__ __launch_bounds__(256)
void attn_window(const float* __restrict__ Q, const float* __restrict__ K,
                 const float* __restrict__ V, float* __restrict__ Oacc)
{
  __shared__ float    Ssc[16][MAXE + 4];   // scores (f32)
  __shared__ _Float16 Pmx[16][MAXE + 8];   // softmax probs (f16), pitch 520h=1040B

  const int lane = threadIdx.x & 31;
  const int wv   = threadIdx.x >> 5;
  const int hi   = lane >> 4;
  const int lo   = lane & 15;
  const int mt   = blockIdx.x;             // 16-row tile inside window
  const int n    = blockIdx.y;             // window index
  const int b    = blockIdx.z >> 3;
  const int h    = blockIdx.z & 7;

  const int s      = n * STR;
  const int decEnd = min(s + WIN, LDEC);
  const int es     = s * 2;                // Le/Ld == 2 exactly
  const int ee     = min(decEnd * 2, LENC);

  const float* Qh = Q + ((size_t)b * CDIM + h * HD) * LDEC;
  const float* Kh = K + ((size_t)b * CDIM + h * HD) * LENC;
  const float* Vh = V + ((size_t)b * CDIM + h * HD) * LENC;

  const int qrow = min(s + mt * 16 + lo, LDEC - 1);

  // ---- phase 1: S = (Q K^T) * hd^-0.5, mask invalid enc cols ----
  for (int nt = wv; nt < MAXE / 16; nt += 8) {
    const int kcol = min(es + nt * 16 + lo, LENC - 1);
    v8f acc = {};
#pragma unroll
    for (int kb = 0; kb < HD; kb += 32) {
      v16h af, bf;
#pragma unroll
      for (int e = 0; e < 16; ++e) {
        af[e] = (_Float16)Qh[(size_t)(kb + a_k(lane, e)) * LDEC + qrow];
        bf[e] = (_Float16)Kh[(size_t)(kb + b_k(lane, e)) * LENC + kcol];
      }
      acc = wmma_f16(af, bf, acc);
    }
#pragma unroll
    for (int p = 0; p < 8; ++p) {
      int row = p + (hi << 3);
      int col = nt * 16 + lo;
      float v = acc[p] * 0.125f;           // 64^-0.5
      if (es + col >= ee) v = -1e30f;
      Ssc[row][col] = v;
    }
  }
  __syncthreads();

  // ---- phase 2: row softmax (16 threads per row, shfl-xor reductions) ----
  {
    const int row = threadIdx.x >> 4;      // 0..15
    const int c0  = threadIdx.x & 15;
    float mx = -3.0e38f;
    for (int j = 0; j < MAXE / 16; ++j) mx = fmaxf(mx, Ssc[row][c0 + j * 16]);
#pragma unroll
    for (int m = 1; m <= 8; m <<= 1) mx = fmaxf(mx, __shfl_xor(mx, m, 32));
    float sum = 0.0f;
    for (int j = 0; j < MAXE / 16; ++j) sum += expf(Ssc[row][c0 + j * 16] - mx);
#pragma unroll
    for (int m = 1; m <= 8; m <<= 1) sum += __shfl_xor(sum, m, 32);
    const float inv = 1.0f / sum;
    for (int j = 0; j < MAXE / 16; ++j)
      Pmx[row][c0 + j * 16] = (_Float16)(expf(Ssc[row][c0 + j * 16] - mx) * inv);
  }
  __syncthreads();

  // ---- phase 3: Wout = P * V (16 x 64), scatter-add valid rows ----
  if (wv < 4) {
    const int dcol = wv * 16 + lo;                 // head dim column
    v8f acc = {};
    for (int kb = 0; kb < MAXE; kb += 32) {
      // A fragment: aligned b128 LDS reads of P
      v8h p0 = *reinterpret_cast<const v8h*>(&Pmx[lo][kb + (hi << 3)]);
      v8h p1 = *reinterpret_cast<const v8h*>(&Pmx[lo][kb + 16 + (hi << 3)]);
      v16h af = cat8(p0, p1);
      v16h bf;
#pragma unroll
      for (int e = 0; e < 16; ++e) {
        int kr = min(es + kb + b_k(lane, e), LENC - 1);
        bf[e] = (_Float16)Vh[(size_t)dcol * LENC + kr];
      }
      acc = wmma_f16(af, bf, acc);
    }
#pragma unroll
    for (int p = 0; p < 8; ++p) {
      int row = p + (hi << 3);
      int g   = s + mt * 16 + row;
      if (g < decEnd)
        atomicAdd(&Oacc[((size_t)b * CDIM + h * HD + dcol) * LDEC + g], acc[p]);
    }
  }
}

// ---------------------------------------------------------------------------
// Host-side orchestration (graph-capture safe: only kernel launches on stream)
// Workspace map (bytes):
//   [0,32M)    bufQ : Q-normed            -> later X (post-wo IN + residual)
//   [32M,96M)  bufK : K-normed            -> later Y2 (wo out) -> H (low half)
//   [96M,160M) bufV : V-normed            -> later H (high half)
//   [160M,192M)bufO : attn accumulator    -> later w2 out
//   [192M,..)  stats (B*H*2 f32) + counts (Ld f32)
// ---------------------------------------------------------------------------
extern "C" void kernel_launch(void* const* d_in, const int* in_sizes, int n_in,
                              void* d_out, int out_size, void* d_ws, size_t ws_size,
                              hipStream_t stream)
{
  const float* xd = (const float*)d_in[0];
  const float* xe = (const float*)d_in[1];
  const float* wq = (const float*)d_in[2];
  const float* bq = (const float*)d_in[3];
  const float* wk = (const float*)d_in[4];
  const float* bk = (const float*)d_in[5];
  const float* wv = (const float*)d_in[6];
  const float* bv = (const float*)d_in[7];
  const float* wo = (const float*)d_in[8];
  const float* bo = (const float*)d_in[9];
  const float* w1 = (const float*)d_in[10];
  const float* b1 = (const float*)d_in[11];
  const float* w2 = (const float*)d_in[12];
  const float* b2 = (const float*)d_in[13];

  char* ws = (char*)d_ws;
  float* bufQ = (float*)(ws);
  float* bufK = (float*)(ws + ((size_t)32  << 20));
  float* bufV = (float*)(ws + ((size_t)96  << 20));
  float* bufO = (float*)(ws + ((size_t)160 << 20));
  float* bufH = bufK;  // 128MB contiguous (K+V regions), valid after attention
  float* stats  = (float*)(ws + ((size_t)192 << 20));
  float* counts = stats + (size_t)BB * HDIM * 2;

  const size_t nQ    = (size_t)BB * CDIM * LDEC;   // 8M elems
  const size_t nK    = (size_t)BB * CDIM * LENC;   // 16M elems
  const size_t statN = (size_t)BB * HDIM * 2;

  auto zero = [&](float* p, size_t n) {
    fill_zero<<<(unsigned)((n + 255) / 256), 256, 0, stream>>>(p, n);
  };

  // ---- Q = IN(wq @ xd + bq) ----
  zero(stats, statN);
  gemm_wmma<<<dim3(LDEC/64, CDIM/128, BB), 256, 0, stream>>>(wq, xd, bq, bufQ, stats, CDIM, CDIM, LDEC, 1);
  fin_norm<<<(unsigned)((nQ + 255)/256), 256, 0, stream>>>(bufQ, stats, nullptr, bufQ, LDEC, nQ);

  // ---- K = IN(wk @ xe + bk) ----
  zero(stats, statN);
  gemm_wmma<<<dim3(LENC/64, CDIM/128, BB), 256, 0, stream>>>(wk, xe, bk, bufK, stats, CDIM, CE, LENC, 1);
  fin_norm<<<(unsigned)((nK + 255)/256), 256, 0, stream>>>(bufK, stats, nullptr, bufK, LENC, nK);

  // ---- V = IN(wv @ xe + bv) ----
  zero(stats, statN);
  gemm_wmma<<<dim3(LENC/64, CDIM/128, BB), 256, 0, stream>>>(wv, xe, bv, bufV, stats, CDIM, CE, LENC, 1);
  fin_norm<<<(unsigned)((nK + 255)/256), 256, 0, stream>>>(bufV, stats, nullptr, bufV, LENC, nK);

  // ---- windowed attention with overlap-average ----
  zero(bufO, nQ);
  counts_k<<<LDEC/256, 256, 0, stream>>>(counts);
  attn_window<<<dim3(WIN/16, NWIN, BB*NH), 256, 0, stream>>>(bufQ, bufK, bufV, bufO);
  div_counts<<<(unsigned)((nQ + 255)/256), 256, 0, stream>>>(bufO, counts, nQ);

  // ---- x = IN(wo @ attn_out + bo) + x_decoder ----
  zero(stats, statN);
  gemm_wmma<<<dim3(LDEC/64, CDIM/128, BB), 256, 0, stream>>>(wo, bufO, bo, bufK, stats, CDIM, CDIM, LDEC, 1);
  fin_norm<<<(unsigned)((nQ + 255)/256), 256, 0, stream>>>(bufK, stats, xd, bufQ, LDEC, nQ);

  // ---- h = gelu(w1 @ x + b1) ----
  gemm_wmma<<<dim3(LDEC/64, HDIM/128, BB), 256, 0, stream>>>(w1, bufQ, b1, bufH, stats, HDIM, CDIM, LDEC, 2);

  // ---- y = IN(w2 @ h + b2) + x ----
  zero(stats, statN);
  gemm_wmma<<<dim3(LDEC/64, CDIM/128, BB), 256, 0, stream>>>(w2, bufH, b2, bufO, stats, CDIM, HDIM, LDEC, 1);
  fin_norm<<<(unsigned)((nQ + 255)/256), 256, 0, stream>>>(bufO, stats, bufQ, (float*)d_out, LDEC, nQ);
}
